// DeltaNet_31877247271520
// MI455X (gfx1250) — compile-verified, matrix operand order checked by reading, not run on hardware
//
#include <hip/hip_runtime.h>

// ---------------------------------------------------------------------------
// Types for CDNA5 WMMA
// ---------------------------------------------------------------------------
typedef __bf16 bf16_t;
typedef __attribute__((ext_vector_type(16))) __bf16 bf16x16;
typedef __attribute__((ext_vector_type(8)))  __bf16 bf16x8;
typedef __attribute__((ext_vector_type(8)))  float  v8f;
typedef __attribute__((ext_vector_type(4)))  float  v4f;
typedef __attribute__((ext_vector_type(2)))  float  v2f;

union FragBF { bf16x16 v; bf16x8 h[2]; };

__device__ __forceinline__ v8f wmma_bf16(bf16x16 a, bf16x16 b, v8f c) {
  return __builtin_amdgcn_wmma_f32_16x16x32_bf16(false, a, false, b, (short)0, c, false, false);
}
__device__ __forceinline__ v8f wmma_f32x4(v2f a, v2f b, v8f c) {
  return __builtin_amdgcn_wmma_f32_16x16x4_f32(false, a, false, b, (short)0, c, false, false);
}

// ---------------------------------------------------------------------------
// gfx1250 async global->LDS copy (ASYNCcnt-tracked), with safe fallback.
// Builtin prototype (from toolchain diagnostics):
//   (AS1 int-vec4* gsrc, AS3 int-vec4* ldst, int offset, int cpol)
// ---------------------------------------------------------------------------
typedef int v4i_t __attribute__((vector_size(16)));
typedef __attribute__((address_space(1))) v4i_t* gvec4_t;
typedef __attribute__((address_space(3))) v4i_t* lvec4_t;

#if defined(__has_builtin)
#if __has_builtin(__builtin_amdgcn_global_load_async_to_lds_b128)
#define HAVE_ASYNC_LD 1
#endif
#endif

__device__ __forceinline__ void async_copy16(const void* g, void* l) {
#ifdef HAVE_ASYNC_LD
  __builtin_amdgcn_global_load_async_to_lds_b128((gvec4_t)g, (lvec4_t)l, 0, 0);
#else
  *(v4f*)l = *(const v4f*)g;
#endif
}
__device__ __forceinline__ void wait_async0() {
#ifdef HAVE_ASYNC_LD
#if __has_builtin(__builtin_amdgcn_s_wait_asynccnt)
  __builtin_amdgcn_s_wait_asynccnt(0);
#else
  asm volatile("s_wait_asynccnt 0x0" ::: "memory");
#endif
#endif
}

// ---------------------------------------------------------------------------
// Problem constants
// ---------------------------------------------------------------------------
#define NB     4
#define NL     4096
#define NDIM   1024
#define NH     4
#define NDK    256
#define NCHK   32
#define NCH    (NL / NCHK)             // 128 chunks
#define NST    5
#define NGIN   (NDIM + 4 * NST * NH)   // 1104
#define NHG    NGIN
#define NBL    (NB * NL)               // 16384 rows
static const size_t NBLD = (size_t)NB * NL * NDIM;  // 16,777,216

// ---------------------------------------------------------------------------
// f32 -> bf16 conversion
// ---------------------------------------------------------------------------
__global__ void cvt_bf16_kernel(const float* __restrict__ in, bf16_t* __restrict__ out, size_t n) {
  size_t i = (size_t)blockIdx.x * 256 + threadIdx.x;
  if (i < n) out[i] = (bf16_t)in[i];
}

// ---------------------------------------------------------------------------
// GEMM: C[M,N] = A[M,K](bf16) * W[N,K]^T(bf16), f32 accumulate via WMMA bf16.
// Block tile 128x64, 8 waves (wave32), wave tile 32x32 (2x2 fragments).
// Double-buffered LDS staging with async global->LDS copies overlapped with
// the current tile's WMMAs.  epi: 0 = none, 1 = bias + exact gelu
// ---------------------------------------------------------------------------
__global__ __launch_bounds__(256) void gemm_bf16_nt(
    const bf16_t* __restrict__ A, const bf16_t* __restrict__ W,
    float* __restrict__ C, int M, int N, int K,
    const float* __restrict__ bias, int epi)
{
  __shared__ bf16_t Asm[2][128][56];   // 32 + 24 pad -> 112B row stride (16B aligned)
  __shared__ bf16_t Bsm[2][64][56];
  const int tid  = threadIdx.x;
  const int lane = tid & 31, wave = tid >> 5;
  const int bm = blockIdx.y * 128, bn = blockIdx.x * 64;
  const int wm = (wave >> 1) * 32, wn = (wave & 1) * 32;
  const int lr = lane & 15, hh = lane >> 4;

  v8f acc[2][2] = {};

  auto stage = [&](int k0, int bufi) {
    // A tile (128x32 bf16): 512 16B chunks over 256 threads
    for (int c2 = tid; c2 < 512; c2 += 256) {
      int r = c2 >> 2, kk = (c2 & 3) * 8;
      int gk = k0 + kk;
      if (gk < K) {
        async_copy16(A + (size_t)(bm + r) * K + gk, &Asm[bufi][r][kk]);
      } else {
        bf16x8 z = {};
        *(bf16x8*)(&Asm[bufi][r][kk]) = z;
      }
    }
    // B tile (64x32 bf16): 256 chunks
    {
      int r = tid >> 2, kk = (tid & 3) * 8;
      int gk = k0 + kk, gn = bn + r;
      if (gn < N && gk < K) {
        async_copy16(W + (size_t)gn * K + gk, &Bsm[bufi][r][kk]);
      } else {
        bf16x8 z = {};
        *(bf16x8*)(&Bsm[bufi][r][kk]) = z;
      }
    }
  };

  const int T = (K + 31) / 32;
  stage(0, 0);
  wait_async0();
  __syncthreads();

  for (int t = 0; t < T; t++) {
    if (t + 1 < T) stage((t + 1) * 32, (t + 1) & 1);   // overlapped with WMMAs below
    const int bi = t & 1;

    FragBF a[2], b[2];
    for (int i = 0; i < 2; i++) {
      const bf16_t* rp = &Asm[bi][wm + i * 16 + lr][0];
      a[i].h[0] = *(const bf16x8*)(rp + hh * 8);
      a[i].h[1] = *(const bf16x8*)(rp + 16 + hh * 8);
      const bf16_t* cp = &Bsm[bi][wn + i * 16 + lr][0];
      b[i].h[0] = *(const bf16x8*)(cp + hh * 8);
      b[i].h[1] = *(const bf16x8*)(cp + 16 + hh * 8);
    }
    for (int i = 0; i < 2; i++)
      for (int j = 0; j < 2; j++)
        acc[i][j] = wmma_bf16(a[i].v, b[j].v, acc[i][j]);

    wait_async0();
    __syncthreads();
  }

  for (int i = 0; i < 2; i++)
    for (int j = 0; j < 2; j++) {
      int col = bn + wn + j * 16 + lr;
      if (col < N) {
        float bv = bias ? bias[col] : 0.f;
        for (int r = 0; r < 8; r++) {
          int row = bm + wm + i * 16 + r + 8 * hh;
          float t = acc[i][j][r] + bv;
          if (epi == 1) t = 0.5f * t * (1.f + erff(t * 0.70710678118654752f));
          C[(size_t)row * N + col] = t;
        }
      }
    }
}

// ---------------------------------------------------------------------------
// Small-N projection: beta = sigmoid(x @ Wb^T), one wave per row, 4 heads
// ---------------------------------------------------------------------------
__global__ __launch_bounds__(256) void beta_kernel(
    const float* __restrict__ x, const float* __restrict__ Wb, float* __restrict__ beta)
{
  int g = blockIdx.x * 8 + (threadIdx.x >> 5);
  int lane = threadIdx.x & 31;
  if (g >= NBL) return;
  size_t xrow = (size_t)g * NDIM;
  for (int h = 0; h < NH; h++) {
    const float* wr = Wb + (size_t)h * NDIM;
    float acc = 0.f;
    for (int k = lane; k < NDIM; k += 32) acc += x[xrow + k] * wr[k];
    for (int m = 1; m < 32; m <<= 1) acc += __shfl_xor(acc, m, 32);
    if (lane == 0) beta[(size_t)g * NH + h] = 1.f / (1.f + expf(-acc));
  }
}

// ---------------------------------------------------------------------------
// depthwise causal conv (K=4) + SiLU
// ---------------------------------------------------------------------------
__global__ void conv_silu_kernel(const float* __restrict__ in, const float* __restrict__ w,
                                 float* __restrict__ out)
{
  size_t idx = (size_t)blockIdx.x * 256 + threadIdx.x;
  if (idx >= NBLD) return;
  int c = (int)(idx % NDIM);
  int t = (int)((idx / NDIM) % NL);
  size_t base = idx - (size_t)c - (size_t)t * NDIM;  // (b,0,0)
  float acc = 0.f;
  for (int j = 0; j < 4; j++) {
    int tt = t - 3 + j;
    if (tt >= 0) acc += in[base + (size_t)tt * NDIM + c] * w[c * 4 + j];
  }
  out[idx] = acc / (1.f + expf(-acc));
}

// ---------------------------------------------------------------------------
// FIR filters (K=3 short, K=31 long) over v streams, per (b,t,h,d)
// ---------------------------------------------------------------------------
__global__ void fir_kernel(const float* __restrict__ v, const float* __restrict__ fS,
                           const float* __restrict__ fL,
                           float* __restrict__ ss, float* __restrict__ sl)
{
  size_t idx = (size_t)blockIdx.x * 256 + threadIdx.x;
  if (idx >= NBLD) return;
  int c = (int)(idx % NDIM);                  // h*256+d
  int t = (int)((idx / NDIM) % NL);
  size_t base = idx - (size_t)c - (size_t)t * NDIM;
  float a0 = 0.f;
  for (int j = 0; j < 3; j++) {
    int tt = t - 2 + j;
    if (tt >= 0) a0 += v[base + (size_t)tt * NDIM + c] * fS[c * 3 + j];
  }
  ss[idx] = a0;
  float a1 = 0.f;
  for (int j = 0; j < 31; j++) {
    int tt = t - 30 + j;
    if (tt >= 0) a1 += v[base + (size_t)tt * NDIM + c] * fL[c * 31 + j];
  }
  sl[idx] = a1;
}

// ---------------------------------------------------------------------------
// Delta-rule chunk preprocessing, one block per (b,h,chunk).
// Produces qn, kn (l2-normalized), u = inv@ (v*beta), w = inv@ (k*beta),
// where inv = (I + strict_lower(kb @ kn^T))^-1 done by forward substitution.
// f32 WMMA 16x16x4 for all matmuls.
// ---------------------------------------------------------------------------
__global__ __launch_bounds__(256) void delta_prep_kernel(
    const float* __restrict__ q, const float* __restrict__ k, const float* __restrict__ v,
    const float* __restrict__ beta,
    float* __restrict__ qn_out, float* __restrict__ kn_out,
    float* __restrict__ u_out, float* __restrict__ w_out)
{
  int blk = blockIdx.x;
  int c  = blk % NCH;
  int h  = (blk / NCH) % NH;
  int b  = blk / (NCH * NH);

  __shared__ float kn[NCHK][NDK + 2];
  __shared__ float kb[NCHK][NDK + 2];
  __shared__ float vb[NCHK][NDK + 2];
  __shared__ float T [NCHK][NCHK + 2];
  __shared__ float sq[NCHK], sk[NCHK], bet[NCHK];

  int tid = threadIdx.x, lane = tid & 31, wave = tid >> 5;
  int lr = lane & 15, hh = lane >> 4;
  size_t rowbase = ((size_t)(b * NL) + (size_t)c * NCHK) * NDIM + (size_t)h * NDK;

  // per-row l2 norms (8 threads per row)
  {
    int r = tid >> 3, j = tid & 7;
    const float* qr = q + rowbase + (size_t)r * NDIM;
    const float* kr = k + rowbase + (size_t)r * NDIM;
    float aq = 0.f, ak = 0.f;
    for (int x = j; x < NDK; x += 8) { float a = qr[x]; aq += a * a; float bb = kr[x]; ak += bb * bb; }
    for (int m = 1; m < 8; m <<= 1) { aq += __shfl_xor(aq, m, 32); ak += __shfl_xor(ak, m, 32); }
    if (j == 0) { sq[r] = rsqrtf(aq + 1e-6f); sk[r] = rsqrtf(ak + 1e-6f); }
    if (tid < NCHK) bet[tid] = beta[((size_t)b * NL + (size_t)c * NCHK + tid) * NH + h];
  }
  __syncthreads();

  // fill LDS tiles + write qn/kn to global
  for (int e = tid; e < NCHK * NDK; e += 256) {
    int r = e >> 8, x = e & 255;
    float qv = q[rowbase + (size_t)r * NDIM + x] * sq[r];
    float kv = k[rowbase + (size_t)r * NDIM + x] * sk[r];
    qn_out[rowbase + (size_t)r * NDIM + x] = qv;
    kn_out[rowbase + (size_t)r * NDIM + x] = kv;
    kn[r][x] = kv;
    kb[r][x] = kv * bet[r];
    vb[r][x] = v[rowbase + (size_t)r * NDIM + x] * bet[r];
  }
  __syncthreads();

  // T = -(kb @ kn^T) masked strictly-lower (waves 0-3, 2x2 tiles, K=256)
  if (wave < 4) {
    int mi = wave >> 1, ni = wave & 1;
    v8f acc = {};
    for (int k0 = 0; k0 < NDK; k0 += 4) {
      v2f a = *(const v2f*)&kb[mi * 16 + lr][k0 + 2 * hh];
      v2f bb = *(const v2f*)&kn[ni * 16 + lr][k0 + 2 * hh];
      acc = wmma_f32x4(a, bb, acc);
    }
    int n = ni * 16 + lr;
    for (int r = 0; r < 8; r++) {
      int m = mi * 16 + r + 8 * hh;
      T[m][n] = (m > n) ? -acc[r] : 0.f;
    }
  }
  __syncthreads();

  // forward substitution (wave 0, lane = column)
  if (wave == 0) {
    int cc = lane;
    for (int i = 1; i < NCHK; i++) {
      float s = 0.f;
      for (int j = 0; j < i; j++) s += T[i][j] * T[j][cc];
      if (cc < i) T[i][cc] += s;
    }
  }
  __syncthreads();

  // u = vb + T@vb ; w = kb + T@kb   (inv = I + T)
  for (int job = wave; job < 64; job += 8) {
    int which = job >> 5;                 // 0 -> u, 1 -> w
    int t2 = job & 31;
    int mi = t2 >> 4, ni = t2 & 15;
    float (*Bm)[NDK + 2] = which ? kb : vb;
    float* outp = which ? w_out : u_out;
    int n = ni * 16 + lr;
    v8f acc = {};
    for (int k0 = 0; k0 < NCHK; k0 += 4) {
      v2f a = *(const v2f*)&T[mi * 16 + lr][k0 + 2 * hh];
      v2f bb; bb[0] = Bm[k0 + 2 * hh][n]; bb[1] = Bm[k0 + 2 * hh + 1][n];
      acc = wmma_f32x4(a, bb, acc);
    }
    size_t obase = (((size_t)(b * NH + h) * NCH + c) * NCHK) * NDK;
    for (int r = 0; r < 8; r++) {
      int m = mi * 16 + r + 8 * hh;
      outp[obase + (size_t)m * NDK + n] = acc[r] + Bm[m][n];
    }
  }
}

// ---------------------------------------------------------------------------
// Delta-rule sequential scan: one block per (b,h,nb), S slice [256 x 64] in LDS.
// For each chunk: u_t = u - w@S ; attn = (qn@kn^T)*incl_lower ;
// o = qn@S + attn@u_t ; S += kn^T @ u_t.   All matmuls via f32 WMMA 16x16x4.
// kn chunk is staged into LDS with async copies overlapped with Phase A.
// ---------------------------------------------------------------------------
#define NSL 64
__global__ __launch_bounds__(256) void delta_scan_kernel(
    const float* __restrict__ qn, const float* __restrict__ kn,
    const float* __restrict__ u, const float* __restrict__ w,
    float* __restrict__ dout)
{
  int blk = blockIdx.x;                   // B*H*4 = 64
  int nb = blk % 4;
  int h  = (blk / 4) % NH;
  int b  = blk / (4 * NH);

  __shared__ float S [NDK][NSL + 2];      // 67.6 KB
  __shared__ float Ut[NCHK][NSL + 2];     // 8.4 KB
  __shared__ float At[NCHK][NCHK + 2];    // 4.3 KB
  __shared__ float Kc[NCHK][NDK + 4];     // 33 KB, 1040B row stride (16B aligned)

  int tid = threadIdx.x, lane = tid & 31, wave = tid >> 5;
  int lr = lane & 15, hh = lane >> 4;

  for (int e = tid; e < NDK * (NSL + 2); e += 256) ((float*)S)[e] = 0.f;
  __syncthreads();

  for (int c = 0; c < NCH; c++) {
    size_t rowbase = ((size_t)(b * NL) + (size_t)c * NCHK) * NDIM + (size_t)h * NDK;
    size_t uwbase  = (((size_t)(b * NH + h) * NCH + c) * NCHK) * NDK;

    // stage kn chunk into LDS (async, 16B chunks), overlapped with Phase A
    for (int e = tid; e < (NCHK * NDK) / 4; e += 256) {
      int r = e >> 6, x = (e & 63) * 4;
      async_copy16(kn + rowbase + (size_t)r * NDIM + x, &Kc[r][x]);
    }

    // Phase A: u_t = u - w@S  (2x4 tiles over 8 waves, K=256)
    {
      int mi = wave >> 2, ni = wave & 3;
      int n = ni * 16 + lr;
      const float* wrow = w + uwbase + (size_t)(mi * 16 + lr) * NDK;
      v8f acc = {};
      for (int k0 = 0; k0 < NDK; k0 += 4) {
        v2f a = *(const v2f*)&wrow[k0 + 2 * hh];
        v2f bb; bb[0] = S[k0 + 2 * hh][n]; bb[1] = S[k0 + 2 * hh + 1][n];
        acc = wmma_f32x4(a, bb, acc);
      }
      const float* ub = u + uwbase + nb * NSL;
      for (int r = 0; r < 8; r++) {
        int m = mi * 16 + r + 8 * hh;
        Ut[m][n] = ub[(size_t)m * NDK + n] - acc[r];
      }
    }
    wait_async0();
    __syncthreads();

    // Phase B: attn = (qn@kn^T) * incl_lower  (waves 0-3, 2x2 tiles, K=256)
    if (wave < 4) {
      int mi = wave >> 1, ni = wave & 1;
      const float* qrow = qn + rowbase + (size_t)(mi * 16 + lr) * NDIM;
      v8f acc = {};
      for (int k0 = 0; k0 < NDK; k0 += 4) {
        v2f a = *(const v2f*)&qrow[k0 + 2 * hh];
        v2f bb = *(const v2f*)&Kc[ni * 16 + lr][k0 + 2 * hh];
        acc = wmma_f32x4(a, bb, acc);
      }
      int n = ni * 16 + lr;
      for (int r = 0; r < 8; r++) {
        int m = mi * 16 + r + 8 * hh;
        At[m][n] = (m >= n) ? acc[r] : 0.f;
      }
    }
    __syncthreads();

    // Phase C: o = qn@S + attn@u_t  (2x4 tiles over 8 waves)
    {
      int mi = wave >> 2, ni = wave & 3;
      int n = ni * 16 + lr;
      const float* qrow = qn + rowbase + (size_t)(mi * 16 + lr) * NDIM;
      v8f acc = {};
      for (int k0 = 0; k0 < NDK; k0 += 4) {
        v2f a = *(const v2f*)&qrow[k0 + 2 * hh];
        v2f bb; bb[0] = S[k0 + 2 * hh][n]; bb[1] = S[k0 + 2 * hh + 1][n];
        acc = wmma_f32x4(a, bb, acc);
      }
      for (int k0 = 0; k0 < NCHK; k0 += 4) {
        v2f a = *(const v2f*)&At[mi * 16 + lr][k0 + 2 * hh];
        v2f bb; bb[0] = Ut[k0 + 2 * hh][n]; bb[1] = Ut[k0 + 2 * hh + 1][n];
        acc = wmma_f32x4(a, bb, acc);
      }
      for (int r = 0; r < 8; r++) {
        int m = mi * 16 + r + 8 * hh;
        dout[rowbase + (size_t)m * NDIM + nb * NSL + n] = acc[r];
      }
    }
    __syncthreads();

    // Phase D: S += kn^T @ u_t  (16x4 tiles, 8 per wave, K=32)
    for (int sub = 0; sub < 2; sub++) {
      int mi = wave * 2 + sub;
      int mrow = mi * 16 + lr;
      for (int ni = 0; ni < 4; ni++) {
        int n = ni * 16 + lr;
        v8f acc;
        for (int r = 0; r < 8; r++) acc[r] = S[mi * 16 + r + 8 * hh][n];
        for (int k0 = 0; k0 < NCHK; k0 += 4) {
          v2f a;  a[0] = Kc[k0 + 2 * hh][mrow]; a[1] = Kc[k0 + 2 * hh + 1][mrow];
          v2f bb; bb[0] = Ut[k0 + 2 * hh][n];   bb[1] = Ut[k0 + 2 * hh + 1][n];
          acc = wmma_f32x4(a, bb, acc);
        }
        for (int r = 0; r < 8; r++) S[mi * 16 + r + 8 * hh][n] = acc[r];
      }
    }
    __syncthreads();
  }
}

// ---------------------------------------------------------------------------
// Per-token stream statistics + gate_in assembly (bf16 for the gate GEMM)
// ---------------------------------------------------------------------------
__global__ __launch_bounds__(256) void stats_kernel(
    const float* __restrict__ x,
    const float* __restrict__ ss, const float* __restrict__ sl,
    const float* __restrict__ dl, const float* __restrict__ vv,
    const float* __restrict__ idv,
    bf16_t* __restrict__ gate_bf)
{
  int bt = blockIdx.x;
  size_t xrow = (size_t)bt * NDIM;
  size_t grow = (size_t)bt * NGIN;
  int tid = threadIdx.x, lane = tid & 31, wave = tid >> 5;

  for (int e = tid; e < NDIM; e += 256) gate_bf[grow + e] = (bf16_t)x[xrow + e];

  const float* bases[5] = { ss, sl, dl, vv, idv };
  for (int job = wave; job < NH * NST; job += 8) {
    int h = job / NST, s = job % NST;
    const float* p = bases[s] + xrow + h * NDK;
    float sx = 0.f, sxx = 0.f, sa = 0.f;
    for (int d = lane; d < NDK; d += 32) { float t = p[d]; sx += t; sxx += t * t; sa += fabsf(t); }
    for (int m = 1; m < 32; m <<= 1) {
      sx += __shfl_xor(sx, m, 32); sxx += __shfl_xor(sxx, m, 32); sa += __shfl_xor(sa, m, 32);
    }
    if (lane == 0) {
      float mean = sx * (1.f / NDK);
      float var  = sxx * (1.f / NDK) - mean * mean;
      float am   = sa * (1.f / NDK);
      float l2   = sqrtf(sxx);
      size_t o = grow + NDIM + (size_t)job * 4;
      gate_bf[o + 0] = (bf16_t)mean; gate_bf[o + 1] = (bf16_t)var;
      gate_bf[o + 2] = (bf16_t)am;   gate_bf[o + 3] = (bf16_t)l2;
    }
  }
}

// ---------------------------------------------------------------------------
// Gate logits + softmax(+eps floor) + stream fusion + id-gate + RMS norm
// One wave per (b,t,h). Writes bf16 normalized output for the final GEMM.
// ---------------------------------------------------------------------------
__global__ __launch_bounds__(256) void fuse_kernel(
    const float* __restrict__ h1, const float* __restrict__ w2, const float* __restrict__ b2,
    const float* __restrict__ ltau, const float* __restrict__ idlogit,
    const float* __restrict__ onw,
    const float* __restrict__ ss, const float* __restrict__ sl,
    const float* __restrict__ dl, const float* __restrict__ vv,
    const float* __restrict__ idv,
    bf16_t* __restrict__ out_bf)
{
  int g = blockIdx.x * 8 + (threadIdx.x >> 5);
  int lane = threadIdx.x & 31;
  int h = g & (NH - 1);
  int bt = g >> 2;
  size_t h1row = (size_t)bt * NHG;

  float itau = expf(-ltau[h]);
  float lg[NST];
  for (int s = 0; s < NST; s++) {
    const float* wr = w2 + (size_t)(h * NST + s) * NHG;
    float acc = 0.f;
    for (int k = lane; k < NHG; k += 32) acc += h1[h1row + k] * wr[k];
    for (int m = 1; m < 32; m <<= 1) acc += __shfl_xor(acc, m, 32);
    lg[s] = (acc + b2[h * NST + s]) * itau;
  }
  float mx = lg[0];
  for (int s = 1; s < NST; s++) mx = fmaxf(mx, lg[s]);
  float es[NST], sum = 0.f;
  for (int s = 0; s < NST; s++) { es[s] = expf(lg[s] - mx); sum += es[s]; }
  float p[NST];
  for (int s = 0; s < NST; s++) p[s] = es[s] / sum * 0.75f + 0.05f;
  float gs = 1.f / (1.f + expf(-idlogit[h]));

  size_t srow = (size_t)bt * NDIM + (size_t)h * NDK;
  float f[8]; float s2 = 0.f;
  for (int i = 0; i < 8; i++) {
    int d = lane + i * 32;
    float val = p[0] * ss[srow + d] + p[1] * sl[srow + d] + p[2] * dl[srow + d]
              + p[3] * vv[srow + d] + (p[4] + gs) * idv[srow + d];
    f[i] = val; s2 += val * val;
  }
  for (int m = 1; m < 32; m <<= 1) s2 += __shfl_xor(s2, m, 32);
  float scale = rsqrtf(s2 * (1.f / NDK) + 1e-5f);
  for (int i = 0; i < 8; i++) {
    int d = lane + i * 32;
    out_bf[srow + d] = (bf16_t)(f[i] * scale * onw[d]);
  }
}

// ---------------------------------------------------------------------------
// Host: workspace carving + launch sequence
// ---------------------------------------------------------------------------
extern "C" void kernel_launch(void* const* d_in, const int* in_sizes, int n_in,
                              void* d_out, int out_size, void* d_ws, size_t ws_size,
                              hipStream_t stream) {
  (void)in_sizes; (void)n_in; (void)out_size; (void)ws_size;
  const float* x    = (const float*)d_in[0];
  const float* Wq   = (const float*)d_in[1];
  const float* Wk   = (const float*)d_in[2];
  const float* Wv   = (const float*)d_in[3];
  const float* Wb   = (const float*)d_in[4];
  const float* cq   = (const float*)d_in[5];
  const float* ck   = (const float*)d_in[6];
  const float* cv   = (const float*)d_in[7];
  const float* firS = (const float*)d_in[8];
  const float* firL = (const float*)d_in[9];
  const float* Wid  = (const float*)d_in[10];
  const float* idlg = (const float*)d_in[11];
  // d_in[12] id_gate_w, d_in[13] id_gate_b unused (untie == 0.0)
  const float* ltau = (const float*)d_in[14];
  const float* fw1  = (const float*)d_in[15];
  const float* fb1  = (const float*)d_in[16];
  const float* fw2  = (const float*)d_in[17];
  const float* fb2  = (const float*)d_in[18];
  const float* onw  = (const float*)d_in[19];
  const float* Wo   = (const float*)d_in[20];
  float* out = (float*)d_out;

  char* ws = (char*)d_ws;
  size_t off = 0;
  auto carve = [&](size_t bytes) -> void* {
    void* p = ws + off;
    off = (off + bytes + 255) & ~(size_t)255;
    return p;
  };

  bf16_t* x_bf   = (bf16_t*)carve(NBLD * 2);
  bf16_t* wq_bf  = (bf16_t*)carve((size_t)NDIM * NDIM * 2);
  bf16_t* wk_bf  = (bf16_t*)carve((size_t)NDIM * NDIM * 2);
  bf16_t* wv_bf  = (bf16_t*)carve((size_t)NDIM * NDIM * 2);
  bf16_t* wid_bf = (bf16_t*)carve((size_t)NDIM * NDIM * 2);
  bf16_t* fw1_bf = (bf16_t*)carve((size_t)NHG * NGIN * 2);
  bf16_t* wo_bf  = (bf16_t*)carve((size_t)NDIM * NDIM * 2);
  float*  qp     = (float*)carve(NBLD * 4);   // proj q, later qn
  float*  kp     = (float*)carve(NBLD * 4);   // proj k, later kn
  float*  vp     = (float*)carve(NBLD * 4);   // proj v, later delta_out
  float*  qc     = (float*)carve(NBLD * 4);   // conv q
  float*  kc     = (float*)carve(NBLD * 4);   // conv k
  float*  vc     = (float*)carve(NBLD * 4);   // conv v
  float*  idv    = (float*)carve(NBLD * 4);
  float*  beta   = (float*)carve((size_t)NBL * NH * 4);
  float*  ubuf   = (float*)carve(NBLD * 4);
  float*  wbuf   = (float*)carve(NBLD * 4);
  float*  ssb    = (float*)carve(NBLD * 4);
  float*  slb    = (float*)carve(NBLD * 4);
  bf16_t* gatebf = (bf16_t*)carve((size_t)NBL * NGIN * 2);
  float*  h1     = (float*)carve((size_t)NBL * NHG * 4);
  bf16_t* fus_bf = (bf16_t*)carve(NBLD * 2);

  const int TPB = 256;
  auto blocksFor = [](size_t n) { return (int)((n + 255) / 256); };

  // bf16 conversions
  cvt_bf16_kernel<<<blocksFor(NBLD), TPB, 0, stream>>>(x, x_bf, NBLD);
  cvt_bf16_kernel<<<blocksFor((size_t)NDIM * NDIM), TPB, 0, stream>>>(Wq, wq_bf, (size_t)NDIM * NDIM);
  cvt_bf16_kernel<<<blocksFor((size_t)NDIM * NDIM), TPB, 0, stream>>>(Wk, wk_bf, (size_t)NDIM * NDIM);
  cvt_bf16_kernel<<<blocksFor((size_t)NDIM * NDIM), TPB, 0, stream>>>(Wv, wv_bf, (size_t)NDIM * NDIM);
  cvt_bf16_kernel<<<blocksFor((size_t)NDIM * NDIM), TPB, 0, stream>>>(Wid, wid_bf, (size_t)NDIM * NDIM);
  cvt_bf16_kernel<<<blocksFor((size_t)NHG * NGIN), TPB, 0, stream>>>(fw1, fw1_bf, (size_t)NHG * NGIN);
  cvt_bf16_kernel<<<blocksFor((size_t)NDIM * NDIM), TPB, 0, stream>>>(Wo, wo_bf, (size_t)NDIM * NDIM);

  // projections (WMMA bf16)
  dim3 g1((NDIM + 63) / 64, NBL / 128);
  gemm_bf16_nt<<<g1, TPB, 0, stream>>>(x_bf, wq_bf, qp, NBL, NDIM, NDIM, nullptr, 0);
  gemm_bf16_nt<<<g1, TPB, 0, stream>>>(x_bf, wk_bf, kp, NBL, NDIM, NDIM, nullptr, 0);
  gemm_bf16_nt<<<g1, TPB, 0, stream>>>(x_bf, wv_bf, vp, NBL, NDIM, NDIM, nullptr, 0);
  gemm_bf16_nt<<<g1, TPB, 0, stream>>>(x_bf, wid_bf, idv, NBL, NDIM, NDIM, nullptr, 0);

  // beta
  beta_kernel<<<NBL / 8, TPB, 0, stream>>>(x, Wb, beta);

  // short convs + silu
  conv_silu_kernel<<<blocksFor(NBLD), TPB, 0, stream>>>(qp, cq, qc);
  conv_silu_kernel<<<blocksFor(NBLD), TPB, 0, stream>>>(kp, ck, kc);
  conv_silu_kernel<<<blocksFor(NBLD), TPB, 0, stream>>>(vp, cv, vc);

  // FIR streams
  fir_kernel<<<blocksFor(NBLD), TPB, 0, stream>>>(vc, firS, firL, ssb, slb);

  // delta-rule chunk prep (WMMA f32) -> qn in qp, kn in kp, u, w
  delta_prep_kernel<<<NB * NH * NCH, TPB, 0, stream>>>(qc, kc, vc, beta, qp, kp, ubuf, wbuf);

  // sequential chunk scan (WMMA f32), delta_out -> vp
  delta_scan_kernel<<<NB * NH * 4, TPB, 0, stream>>>(qp, kp, ubuf, wbuf, vp);

  // stats + gate input (bf16)
  stats_kernel<<<NBL, TPB, 0, stream>>>(x, ssb, slb, vp, vc, idv, gatebf);

  // gate MLP layer 1: gelu(gate_in @ fg_w1^T + b1)
  dim3 g2((NHG + 63) / 64, NBL / 128);
  gemm_bf16_nt<<<g2, TPB, 0, stream>>>(gatebf, fw1_bf, h1, NBL, NHG, NGIN, fb1, 1);

  // logits + softmax + fusion + RMS norm -> bf16
  fuse_kernel<<<(NBL * NH) / 8, TPB, 0, stream>>>(h1, fw2, fb2, ltau, idlg, onw,
                                                  ssb, slb, vp, vc, idv, fus_bf);

  // final projection
  gemm_bf16_nt<<<g1, TPB, 0, stream>>>(fus_bf, wo_bf, out, NBL, NDIM, NDIM, nullptr, 0);
}
